// GraphSAGEAutoencoder_77421080477957
// MI455X (gfx1250) — matrix operand order repaired and checked by reading, not compiled, not run on hardware
//
#include <hip/hip_runtime.h>

#define NODELEN 17949
#define NNODES  512
#define NEDGES  4096
#define H2DIM   768
#define EMBDIM  128
#define INDIM   (2 * NODELEN)      // 35898
#define EKP     35904              // INDIM padded to multiple of 32

typedef __attribute__((ext_vector_type(16))) _Float16 v16h;
typedef __attribute__((ext_vector_type(8)))  _Float16 v8h;
typedef __attribute__((ext_vector_type(8)))  float    v8f;
typedef __attribute__((ext_vector_type(2)))  float    f32x2;

// ---------------- aggregation kernels ----------------

__global__ void zero_f32_kernel(float* __restrict__ p, long long n) {
    long long i = (long long)blockIdx.x * blockDim.x + threadIdx.x;
    long long stride = (long long)gridDim.x * blockDim.x;
    for (; i < n; i += stride) p[i] = 0.0f;
}

__global__ void degree_kernel(const int* __restrict__ dst, float* __restrict__ deg, int E) {
    int e = blockIdx.x * blockDim.x + threadIdx.x;
    if (e < E) atomicAdd(&deg[dst[e]], 1.0f);
}

__global__ void scatter_add_kernel(const float* __restrict__ feat, float* __restrict__ out,
                                   const int* __restrict__ esrc, const int* __restrict__ edst,
                                   int F) {
    int f = blockIdx.x * blockDim.x + threadIdx.x;
    int e = blockIdx.y;
    if (f >= F) return;
    int s = esrc[e];
    int d = edst[e];
    atomicAdd(&out[(size_t)d * F + f], feat[(size_t)s * F + f]);
}

// h1: divide by max(deg,1), zero col 0, in place (f32)
__global__ void seg_mean_fin_kernel(float* __restrict__ h, const float* __restrict__ deg, int F) {
    int f = blockIdx.x * blockDim.x + threadIdx.x;
    int n = blockIdx.y;
    if (f >= F) return;
    float d = deg[n];
    d = d > 1.0f ? d : 1.0f;
    float v = h[(size_t)n * F + f] / d;
    h[(size_t)n * F + f] = (f == 0) ? 0.0f : v;
}

// embh[:, 0:NODELEN] = f16(x) with col 0 zeroed
__global__ void pack_x_f16_kernel(const float* __restrict__ x, _Float16* __restrict__ embh) {
    int f = blockIdx.x * blockDim.x + threadIdx.x;
    int n = blockIdx.y;
    if (f >= NODELEN) return;
    float v = (f == 0) ? 0.0f : x[(size_t)n * NODELEN + f];
    embh[(size_t)n * EKP + f] = (_Float16)v;
}

// embh[:, NODELEN:EKP] = f16(h2acc/deg) with col 0 zeroed, plus zero K-padding
__global__ void fin_h2_f16_kernel(const float* __restrict__ h2acc, const float* __restrict__ deg,
                                  _Float16* __restrict__ embh) {
    int f = blockIdx.x * blockDim.x + threadIdx.x;     // 0 .. EKP-NODELEN-1
    int n = blockIdx.y;
    const int HP = EKP - NODELEN;                      // NODELEN + 6
    if (f >= HP) return;
    float v = 0.0f;
    if (f < NODELEN && f != 0) {
        float d = deg[n];
        d = d > 1.0f ? d : 1.0f;
        v = h2acc[(size_t)n * NODELEN + f] / d;
    }
    embh[(size_t)n * EKP + NODELEN + f] = (_Float16)v;
}

// ---------------- WMMA GEMM, 2x2 register-tiled ----------------
// C = act( Ah[M,Kp](f16, Kp%32==0, zero-padded) @ B[Ntot,K](f32)^T + bias )
// One wave computes a 32x32 output supertile = 4 WMMA tiles with 4 independent
// accumulators; per 32-K step: 4 b128 A-loads + 8 b128 B-loads -> 4 WMMAs.
// Hot loop is guard-free (B rows clamped; junk columns dropped at store time).
// K tail (<32, GEMM1 only) is one guarded epilogue step.

__global__ __launch_bounds__(256) void gemm_wmma_kernel(
    const _Float16* __restrict__ Ah, int Kp,
    const float* __restrict__ B, int K,
    const float* __restrict__ bias,
    float* __restrict__ Cf, _Float16* __restrict__ Ch,
    int M, int Ntot, int do_relu)
{
    const int lane = threadIdx.x & 31;
    const int wave = threadIdx.x >> 5;
    const int Mt2 = M >> 5;                     // 32-row supertiles (M % 32 == 0)
    const int Nt2 = (Ntot + 31) >> 5;
    const int tile = blockIdx.x * 8 + wave;
    if (tile >= Mt2 * Nt2) return;              // wave-uniform: EXEC full at WMMA
    const int tile_m = (tile % Mt2) << 5;       // M fastest: waves in a block share B
    const int tile_n = (tile / Mt2) << 5;

    // A: lane owns row M = lane&15; K runs split by lane half (contiguous k = 16*half + s, same order as B)
    const int a_koff = (lane >> 4) << 4;        // +0 / +16
    const int a_row0 = tile_m + (lane & 15);
    // B: lane owns col N = lane&15; K base split by lane half
    const int b_koff = (lane >> 4) << 4;        // +0 / +16
    int b_col0 = tile_n + (lane & 15);
    int b_col1 = b_col0 + 16;
    if (b_col0 >= Ntot) b_col0 = Ntot - 1;      // clamp: junk cols never stored
    if (b_col1 >= Ntot) b_col1 = Ntot - 1;

    const _Float16* __restrict__ arow0 = Ah + (size_t)a_row0 * Kp + a_koff;
    const _Float16* __restrict__ arow1 = arow0 + (size_t)16 * Kp;
    const float*    __restrict__ brow0 = B + (size_t)b_col0 * K + b_koff;
    const float*    __restrict__ brow1 = B + (size_t)b_col1 * K + b_koff;

    v8f acc00 = {}, acc01 = {}, acc10 = {}, acc11 = {};

    auto loadA = [&](const _Float16* ar, int kb) -> v16h {
        v8h lo = *(const v8h*)(ar + kb);          // 16B-aligned b128
        v8h hi = *(const v8h*)(ar + kb + 8);
        return __builtin_shufflevector(lo, hi,
               0, 1, 2, 3, 4, 5, 6, 7, 8, 9, 10, 11, 12, 13, 14, 15);
    };
    auto loadB = [&](const float* br, int kb) -> v16h {
        v16h b;
        const f32x2* bp = (const f32x2*)(br + kb);
        #pragma unroll
        for (int j = 0; j < 8; ++j) {
            f32x2 t = bp[j];
            b[2 * j]     = (_Float16)t.x;
            b[2 * j + 1] = (_Float16)t.y;
        }
        return b;
    };
    #define WMMA_STEP(a_, b_, c_) __builtin_amdgcn_wmma_f32_16x16x32_f16( \
            false, (a_), false, (b_), (short)0, (c_), false, false)

    int kk = 0;
    for (; kk + 32 <= K; kk += 32) {            // guard-free hot loop
        __builtin_prefetch((const void*)(brow0 + kk + 128), 0, 1);
        __builtin_prefetch((const void*)(brow1 + kk + 128), 0, 1);
        v16h a0 = loadA(arow0, kk);
        v16h a1 = loadA(arow1, kk);
        v16h b0 = loadB(brow0, kk);
        v16h b1 = loadB(brow1, kk);
        acc00 = WMMA_STEP(a0, b0, acc00);       // 4 independent WMMAs
        acc01 = WMMA_STEP(a0, b1, acc01);
        acc10 = WMMA_STEP(a1, b0, acc10);
        acc11 = WMMA_STEP(a1, b1, acc11);
    }

    if (kk < K) {                               // K tail (GEMM1: 26 elems), runs once
        v16h a0 = loadA(arow0, kk);             // A zero-padded up to Kp
        v16h a1 = loadA(arow1, kk);
        v16h b0, b1;
        #pragma unroll
        for (int j = 0; j < 16; ++j) {
            int kabs = kk + b_koff + j;
            b0[j] = (_Float16)((kabs < K) ? B[(size_t)b_col0 * K + kabs] : 0.0f);
            b1[j] = (_Float16)((kabs < K) ? B[(size_t)b_col1 * K + kabs] : 0.0f);
        }
        acc00 = WMMA_STEP(a0, b0, acc00);
        acc01 = WMMA_STEP(a0, b1, acc01);
        acc10 = WMMA_STEP(a1, b0, acc10);
        acc11 = WMMA_STEP(a1, b1, acc11);
    }
    #undef WMMA_STEP

    // C/D layout: VGPR r, lanes 0-15 -> M=r, lanes 16-31 -> M=r+8; N = lane&15
    const int om = (lane >> 4) << 3;
    const int on = lane & 15;
    auto store_tile = [&](const v8f& acc, int m_base, int n_base) {
        const int out_n = n_base + on;
        if (out_n < Ntot) {
            const float bv = bias[out_n];
            #pragma unroll
            for (int r = 0; r < 8; ++r) {
                float v = acc[r] + bv;
                if (do_relu) v = v > 0.0f ? v : 0.0f;
                const size_t idx = (size_t)(m_base + om + r) * Ntot + out_n;
                if (Cf) Cf[idx] = v;
                if (Ch) Ch[idx] = (_Float16)v;
            }
        }
    };
    store_tile(acc00, tile_m,      tile_n);
    store_tile(acc01, tile_m,      tile_n + 16);
    store_tile(acc10, tile_m + 16, tile_n);
    store_tile(acc11, tile_m + 16, tile_n + 16);
}

// ---------------- launcher ----------------

extern "C" void kernel_launch(void* const* d_in, const int* in_sizes, int n_in,
                              void* d_out, int out_size, void* d_ws, size_t ws_size,
                              hipStream_t stream) {
    const float* x    = (const float*)d_in[0];
    const int*   esrc = (const int*)  d_in[1];
    const int*   edst = (const int*)  d_in[2];
    const float* W_e2 = (const float*)d_in[3];
    const float* b_e2 = (const float*)d_in[4];
    const float* W_e3 = (const float*)d_in[5];
    const float* b_e3 = (const float*)d_in[6];
    const float* W_d1 = (const float*)d_in[7];
    const float* b_d1 = (const float*)d_in[8];
    const float* W_d3 = (const float*)d_in[9];
    const float* b_d3 = (const float*)d_in[10];

    const size_t FEAT_BYTES = (size_t)NNODES * NODELEN * sizeof(float);      // 36,759,552
    const size_t EMBH_BYTES = (size_t)NNODES * EKP * sizeof(_Float16);       // 36,765,696
    const size_t HIDH_BYTES = (size_t)NNODES * H2DIM * sizeof(_Float16);     //    786,432

    char* ws = (char*)d_ws;
    float*    deg   = (float*)(ws + 0);
    // h1 (f32 hop-1 result) and embh (f16 packed A for GEMM1) share one slot:
    // h1 is dead after the hop-2 scatter, which precedes all embh writes.
    char*     slot0 = ws + 4096;                         // size = EMBH_BYTES (> FEAT_BYTES)
    float*    h1    = (float*)slot0;
    _Float16* embh  = (_Float16*)slot0;
    float*    h2acc = (float*)(ws + 4096 + EMBH_BYTES);
    _Float16* hb1h  = (_Float16*)(ws + 4096 + EMBH_BYTES + FEAT_BYTES);
    _Float16* hb2h  = (_Float16*)(ws + 4096 + EMBH_BYTES + FEAT_BYTES + HIDH_BYTES);
    _Float16* ench  = (_Float16*)(ws + 4096 + EMBH_BYTES + FEAT_BYTES + 2 * HIDH_BYTES);

    float* enc = (float*)d_out;                      // [512,128]
    float* dec = enc + (size_t)NNODES * EMBDIM;      // [512,35898]

    const dim3 blk(256);
    const dim3 gEF((NODELEN + 255) / 256, NEDGES);
    const dim3 gNF((NODELEN + 255) / 256, NNODES);
    const dim3 gNH((EKP - NODELEN + 255) / 256, NNODES);

    // hop 1: h1 = seg_mean(x), col0 = 0
    zero_f32_kernel<<<2048, blk, 0, stream>>>(h1, (long long)NNODES * NODELEN);
    zero_f32_kernel<<<1, blk, 0, stream>>>(deg, (long long)NNODES);
    degree_kernel<<<(NEDGES + 255) / 256, blk, 0, stream>>>(edst, deg, NEDGES);
    scatter_add_kernel<<<gEF, blk, 0, stream>>>(x, h1, esrc, edst, NODELEN);
    seg_mean_fin_kernel<<<gNF, blk, 0, stream>>>(h1, deg, NODELEN);

    // hop 2: h2acc = scatter(h1); (h1 dead after this point)
    zero_f32_kernel<<<2048, blk, 0, stream>>>(h2acc, (long long)NNODES * NODELEN);
    scatter_add_kernel<<<gEF, blk, 0, stream>>>(h1, h2acc, esrc, edst, NODELEN);

    // pack emb = [x | mean(h2acc)] as f16, cols 0 and NODELEN zeroed, K-padded
    pack_x_f16_kernel<<<gNF, blk, 0, stream>>>(x, embh);
    fin_h2_f16_kernel<<<gNH, blk, 0, stream>>>(h2acc, deg, embh);

    auto gemm = [&](const _Float16* Ah, int Kp, const float* B, int K,
                    const float* bias, float* Cf, _Float16* Ch, int N, int relu) {
        int Mt2 = NNODES / 32;
        int Nt2 = (N + 31) / 32;
        int blocks = (Mt2 * Nt2 + 7) / 8;        // 8 waves per 256-thread block
        gemm_wmma_kernel<<<blocks, blk, 0, stream>>>(Ah, Kp, B, K, bias, Cf, Ch,
                                                     NNODES, N, relu);
    };

    // hb1 = relu(emb @ W_e2^T + b_e2)           -> f16 only (feeds GEMM2)
    gemm(embh, EKP,   W_e2, INDIM,  b_e2, nullptr, hb1h, H2DIM,  1);
    // enc = hb1 @ W_e3^T + b_e3                 -> f32 (output) + f16 (feeds GEMM3)
    gemm(hb1h, H2DIM, W_e3, H2DIM,  b_e3, enc,     ench, EMBDIM, 0);
    // hb2 = relu(enc @ W_d1^T + b_d1)           -> f16 only (feeds GEMM4)
    gemm(ench, EMBDIM,W_d1, EMBDIM, b_d1, nullptr, hb2h, H2DIM,  1);
    // dec = hb2 @ W_d3^T + b_d3                 -> f32 output
    gemm(hb2h, H2DIM, W_d3, H2DIM,  b_d3, dec,     nullptr, INDIM, 0);
}
